// SelfAttention_v2_16458314678750
// MI455X (gfx1250) — matrix operand order repaired and checked by reading, not compiled, not run on hardware
//
#include <hip/hip_runtime.h>

// ---------------------------------------------------------------------------
// Self-attention (N=4096, D=2048) for MI455X / gfx1250, wave32 + WMMA +
// gfx1250 async global->LDS copies (ASYNCcnt).
// Pipeline:
//   1) f32->bf16 convert x;  f32->bf16 convert+TRANSPOSE W_q/W_k/W_v
//   2) Q,K,V = X @ W   (B given as B^T -> pure-copy LDS fill, async)
//   3) S = sqrt(Q @ K^T)  (K row-major IS B^T -> async fill)
//   4) P = row softmax(S) (bf16)
//   5) out = P @ V  (B normal -> sync transpose fill for B, async for A)
// Workspace (184 MB):
//   Xb 16MB | WqT 8MB | WkT 8MB | WvT 8MB | Qb 16MB | Kb 16MB | Vb 16MB
//   | S (f32) 64MB | P (bf16) 32MB
// ---------------------------------------------------------------------------

#define N_TOK   4096
#define D_MODEL 2048

typedef __bf16 bf16;
typedef __attribute__((ext_vector_type(16))) __bf16 v16bf;
typedef __attribute__((ext_vector_type(8)))  __bf16 v8bf;
typedef __attribute__((ext_vector_type(8)))  float  v8f;

static __device__ __forceinline__ unsigned short f32_to_bf16_bits(float f) {
    unsigned u = __float_as_uint(f);
    u += 0x7FFFu + ((u >> 16) & 1u);       // round-to-nearest-even
    return (unsigned short)(u >> 16);
}
static __device__ __forceinline__ bf16 u16_as_bf16(unsigned short s) {
    union { unsigned short u; bf16 b; } x; x.u = s; return x.b;
}
static __device__ __forceinline__ unsigned short bf16_as_u16(bf16 b) {
    union { bf16 b; unsigned short u; } x; x.b = b; return x.u;
}

// ---- gfx1250 async global->LDS copy (tracked by ASYNCcnt) -----------------
// VDST VGPR holds the LDS byte address (generic LDS ptr low 32 bits == group
// segment offset; flat->LDS mapping uses addr[31:0] per CDNA5 ISA 10.2).
static __device__ __forceinline__ void async_copy_b128(void* lds, const void* gptr) {
    asm volatile("global_load_async_to_lds_b128 %0, %1, off"
                 :: "v"((unsigned)(unsigned long long)lds),
                    "v"((unsigned long long)gptr)
                 : "memory");
}
static __device__ __forceinline__ void wait_async0() {
    asm volatile("s_wait_asynccnt 0" ::: "memory");
}

// ---------------------------------------------------------------------------
// Elementwise f32 -> bf16 (vectorized x4)
// ---------------------------------------------------------------------------
__global__ __launch_bounds__(256)
void cvt_f32_bf16(const float* __restrict__ in, bf16* __restrict__ out, int n) {
    int i = (blockIdx.x * 256 + threadIdx.x) * 4;
    if (i + 3 < n) {
        float4 v = *reinterpret_cast<const float4*>(in + i);
        union { unsigned short s[4]; uint2 u; } p;
        p.s[0] = f32_to_bf16_bits(v.x);
        p.s[1] = f32_to_bf16_bits(v.y);
        p.s[2] = f32_to_bf16_bits(v.z);
        p.s[3] = f32_to_bf16_bits(v.w);
        *reinterpret_cast<uint2*>(out + i) = p.u;
    }
}

// ---------------------------------------------------------------------------
// f32 [n x n] -> bf16 transposed [n x n]: out[c][r] = (bf16)in[r][c].
// 64x64 tiles staged through LDS; coalesced read and write.
// ---------------------------------------------------------------------------
__global__ __launch_bounds__(256)
void cvt_transpose_f32_bf16(const float* __restrict__ in, bf16* __restrict__ out, int n)
{
    __shared__ __attribute__((aligned(16))) bf16 tile[64][72];
    const int tid = threadIdx.x;
    const int r0 = blockIdx.y * 64;    // source row block
    const int c0 = blockIdx.x * 64;    // source col block
    #pragma unroll
    for (int p = 0; p < 4; ++p) {
        int row = p * 16 + (tid >> 4);
        int col = (tid & 15) * 4;
        float4 v = *reinterpret_cast<const float4*>(in + (size_t)(r0 + row) * n + c0 + col);
        tile[row][col + 0] = u16_as_bf16(f32_to_bf16_bits(v.x));
        tile[row][col + 1] = u16_as_bf16(f32_to_bf16_bits(v.y));
        tile[row][col + 2] = u16_as_bf16(f32_to_bf16_bits(v.z));
        tile[row][col + 3] = u16_as_bf16(f32_to_bf16_bits(v.w));
    }
    __syncthreads();
    #pragma unroll
    for (int p = 0; p < 4; ++p) {
        int c = p * 16 + (tid >> 4);       // output row = source column
        int r = (tid & 15) * 4;            // output col = source row
        union { unsigned short s[4]; uint2 u; } pk;
        pk.s[0] = bf16_as_u16(tile[r + 0][c]);
        pk.s[1] = bf16_as_u16(tile[r + 1][c]);
        pk.s[2] = bf16_as_u16(tile[r + 2][c]);
        pk.s[3] = bf16_as_u16(tile[r + 3][c]);
        *reinterpret_cast<uint2*>(out + (size_t)(c0 + c) * n + r0 + r) = pk.u;
    }
}

// ---------------------------------------------------------------------------
// bf16 GEMM: C[M,N] = A[M,K] * B, f32 accumulate via v_wmma_f32_16x16x32_bf16.
//   BT=true : B given row-major [N,K] (=B^T) -> LDS fill is a pure async copy
//   BT=false: B row-major [K,N] -> sync transpose fill (P@V only)
//   OUT_MODE: 0 = f32, 1 = f32 with fused sqrt, 2 = bf16
// Block tile 128x128, BK=32, 256 threads = 8 waves (2x4), wave tile 64x32.
// A tiles always filled with global_load_async_to_lds_b128 (overlaps WMMA).
// ---------------------------------------------------------------------------
constexpr int BM = 128, BN = 128, BK = 32, BKP = 40; // BKP: padded LDS stride

template<bool BT, int OUT_MODE>
__global__ __launch_bounds__(256)
void gemm_wmma(const bf16* __restrict__ A, const bf16* __restrict__ B,
               void* __restrict__ Cout, int M, int N, int K)
{
    __shared__ __attribute__((aligned(16))) bf16 As[2][BM][BKP];
    __shared__ __attribute__((aligned(16))) bf16 Bs[2][BN][BKP]; // [n][k]
    (void)M;

    const int tid  = threadIdx.x;
    const int lane = tid & 31;          // wave32
    const int wave = tid >> 5;
    const int wm   = wave >> 2;         // 0..1  (M direction)
    const int wn   = wave & 3;          // 0..3  (N direction)
    const int half = lane >> 4;         // K-half selector per ISA layout
    const int l16  = lane & 15;

    const int m0 = blockIdx.y * BM;
    const int n0 = blockIdx.x * BN;

    v8f acc[4][2];
    #pragma unroll
    for (int i = 0; i < 4; ++i)
        #pragma unroll
        for (int j = 0; j < 2; ++j) { v8f z = {}; acc[i][j] = z; }

    const int kTiles = K / BK;

    // Issue the LDS fill for tile kt into buffer `buf`.
    auto issueTile = [&](int kt, int buf) {
        const int k0 = kt * BK;
        // ---- A tile: 128x32 bf16 = 512 x 16B chunks, 2 per thread, async --
        for (int s = 0; s < 2; ++s) {
            int c   = tid + s * 256;
            int row = c >> 2;
            int off = (c & 3) * 8;
            async_copy_b128(&As[buf][row][off],
                            A + (size_t)(m0 + row) * K + k0 + off);
        }
        if (BT) {
            // B^T given: rows are N, contiguous in K -> pure async copy
            for (int s = 0; s < 2; ++s) {
                int c   = tid + s * 256;
                int row = c >> 2;
                int off = (c & 3) * 8;
                async_copy_b128(&Bs[buf][row][off],
                                B + (size_t)(n0 + row) * K + k0 + off);
            }
        } else {
            // B row-major [K,N]: read 2 k-rows x 8 n, transpose-pack b32 stores
            int k2 = (tid >> 4) * 2;        // 0,2,...,30
            int n8 = (tid & 15) * 8;        // 0,8,...,120
            v8bf r0 = *reinterpret_cast<const v8bf*>(B + (size_t)(k0 + k2)     * N + n0 + n8);
            v8bf r1 = *reinterpret_cast<const v8bf*>(B + (size_t)(k0 + k2 + 1) * N + n0 + n8);
            #pragma unroll
            for (int j = 0; j < 8; ++j) {
                union { bf16 h[2]; unsigned u; } p;
                p.h[0] = r0[j]; p.h[1] = r1[j];
                *reinterpret_cast<unsigned*>(&Bs[buf][n8 + j][k2]) = p.u;
            }
        }
    };

    issueTile(0, 0);
    wait_async0();
    __syncthreads();

    for (int kt = 0; kt < kTiles; ++kt) {
        const int cur = kt & 1;
        // Prefetch next tile: async copies overlap the WMMA block below.
        if (kt + 1 < kTiles) issueTile(kt + 1, cur ^ 1);
        if (kt + 2 < kTiles)  // gfx1250 global_prefetch_b8, two tiles ahead
            __builtin_prefetch(A + (size_t)(m0 + (tid >> 1)) * K + (kt + 2) * BK, 0, 1);

        // Fragment loads follow the CDNA5 16-bit WMMA VGPR striping:
        //   A 16x32: lanes 0-15 hold {K0-7, K16-23}, lanes 16-31 {K8-15, K24-31}
        //   B 32x16: lanes 0-15 hold {K0-7, K16-23} of col l16, lanes 16-31 {K8-15, K24-31}
        union Frag { v16bf v; v8bf h[2]; };
        Frag a[4], b[2];
        #pragma unroll
        for (int i = 0; i < 4; ++i) {
            int row = wm * 64 + i * 16 + l16;
            a[i].h[0] = *reinterpret_cast<const v8bf*>(&As[cur][row][8 * half]);
            a[i].h[1] = *reinterpret_cast<const v8bf*>(&As[cur][row][16 + 8 * half]);
        }
        #pragma unroll
        for (int j = 0; j < 2; ++j) {
            int col = wn * 32 + j * 16 + l16;
            b[j].h[0] = *reinterpret_cast<const v8bf*>(&Bs[cur][col][8 * half]);
            b[j].h[1] = *reinterpret_cast<const v8bf*>(&Bs[cur][col][16 + 8 * half]);
        }
        #pragma unroll
        for (int i = 0; i < 4; ++i)
            #pragma unroll
            for (int j = 0; j < 2; ++j)
                acc[i][j] = __builtin_amdgcn_wmma_f32_16x16x32_bf16(
                    false, a[i].v, false, b[j].v, (short)0, acc[i][j], false, false);

        wait_async0();       // this wave's async fills of `nxt` landed in LDS
        __syncthreads();     // all waves' fills visible
    }

    // Epilogue. C/D layout: VGPR r -> M = r + 8*half, N = lane&15.
    #pragma unroll
    for (int i = 0; i < 4; ++i) {
        int rowBase = m0 + wm * 64 + i * 16 + half * 8;
        #pragma unroll
        for (int j = 0; j < 2; ++j) {
            int col = n0 + wn * 32 + j * 16 + l16;
            #pragma unroll
            for (int r = 0; r < 8; ++r) {
                float v = acc[i][j][r];
                size_t idx = (size_t)(rowBase + r) * N + col;
                if (OUT_MODE == 0)      ((float*)Cout)[idx] = v;
                else if (OUT_MODE == 1) ((float*)Cout)[idx] = sqrtf(v);
                else                    ((bf16*)Cout)[idx]  = u16_as_bf16(f32_to_bf16_bits(v));
            }
        }
    }
}

// ---------------------------------------------------------------------------
// Row softmax: one block (8 waves) per row of S[4096]. exp row cached in LDS.
// ---------------------------------------------------------------------------
__global__ __launch_bounds__(256)
void softmax_row_kernel(const float* __restrict__ S, bf16* __restrict__ P, int n)
{
    const int row = blockIdx.x;
    const float* s = S + (size_t)row * n;
    __shared__ float buf[N_TOK];
    __shared__ float red[8];
    const int tid = threadIdx.x, lane = tid & 31, wave = tid >> 5;

    float mx = -3.402823466e38f;
    for (int c = tid; c < n; c += 256) {
        float v = s[c];
        buf[c] = v;
        mx = fmaxf(mx, v);
    }
    for (int off = 16; off; off >>= 1) mx = fmaxf(mx, __shfl_xor(mx, off, 32));
    if (lane == 0) red[wave] = mx;
    __syncthreads();
    if (wave == 0) {
        float m = (lane < 8) ? red[lane] : -3.402823466e38f;
        for (int off = 4; off; off >>= 1) m = fmaxf(m, __shfl_xor(m, off, 32));
        if (lane == 0) red[0] = m;
    }
    __syncthreads();
    mx = red[0];

    float sum = 0.f;
    for (int c = tid; c < n; c += 256) {
        float e = __expf(buf[c] - mx);
        buf[c] = e;
        sum += e;
    }
    __syncthreads();
    for (int off = 16; off; off >>= 1) sum += __shfl_xor(sum, off, 32);
    if (lane == 0) red[wave] = sum;
    __syncthreads();
    if (wave == 0) {
        float t = (lane < 8) ? red[lane] : 0.f;
        for (int off = 4; off; off >>= 1) t += __shfl_xor(t, off, 32);
        if (lane == 0) red[0] = t;
    }
    __syncthreads();
    const float inv = 1.0f / red[0];
    for (int c = tid; c < n; c += 256)
        P[(size_t)row * n + c] = u16_as_bf16(f32_to_bf16_bits(buf[c] * inv));
}

// ---------------------------------------------------------------------------
extern "C" void kernel_launch(void* const* d_in, const int* in_sizes, int n_in,
                              void* d_out, int out_size, void* d_ws, size_t ws_size,
                              hipStream_t stream)
{
    (void)in_sizes; (void)n_in; (void)out_size; (void)ws_size;
    const float* x  = (const float*)d_in[0];
    const float* Wq = (const float*)d_in[1];
    const float* Wk = (const float*)d_in[2];
    const float* Wv = (const float*)d_in[3];
    float* out = (float*)d_out;

    const int N = N_TOK, D = D_MODEL;
    char* ws = (char*)d_ws;
    size_t off = 0;
    bf16* Xb  = (bf16*)(ws + off); off += (size_t)N * D * sizeof(bf16);
    bf16* WqT = (bf16*)(ws + off); off += (size_t)D * D * sizeof(bf16);
    bf16* WkT = (bf16*)(ws + off); off += (size_t)D * D * sizeof(bf16);
    bf16* WvT = (bf16*)(ws + off); off += (size_t)D * D * sizeof(bf16);
    bf16* Qb  = (bf16*)(ws + off); off += (size_t)N * D * sizeof(bf16);
    bf16* Kb  = (bf16*)(ws + off); off += (size_t)N * D * sizeof(bf16);
    bf16* Vb  = (bf16*)(ws + off); off += (size_t)N * D * sizeof(bf16);
    float* S  = (float*)(ws + off); off += (size_t)N * N * sizeof(float);
    bf16* P   = (bf16*)(ws + off); off += (size_t)N * N * sizeof(bf16);

    // 1) convert x; convert+transpose weights (so Q/K/V GEMMs take the
    //    pure-copy async-LDS path).
    cvt_f32_bf16<<<dim3((N * D) / 1024), dim3(256), 0, stream>>>(x, Xb, N * D);
    cvt_transpose_f32_bf16<<<dim3(D / 64, D / 64), dim3(256), 0, stream>>>(Wq, WqT, D);
    cvt_transpose_f32_bf16<<<dim3(D / 64, D / 64), dim3(256), 0, stream>>>(Wk, WkT, D);
    cvt_transpose_f32_bf16<<<dim3(D / 64, D / 64), dim3(256), 0, stream>>>(Wv, WvT, D);

    const dim3 blk(256);
    // 2) Q,K,V = X @ W   (W^T supplied -> BT path, bf16 out)
    gemm_wmma<true, 2><<<dim3(D / BN, N / BM), blk, 0, stream>>>(Xb, WqT, Qb, N, D, D);
    gemm_wmma<true, 2><<<dim3(D / BN, N / BM), blk, 0, stream>>>(Xb, WkT, Kb, N, D, D);
    gemm_wmma<true, 2><<<dim3(D / BN, N / BM), blk, 0, stream>>>(Xb, WvT, Vb, N, D, D);
    // 3) S = sqrt(Q @ K^T)  (K row-major IS B^T)
    gemm_wmma<true, 1><<<dim3(N / BN, N / BM), blk, 0, stream>>>(Qb, Kb, S, N, N, D);
    // 4) P = softmax(S) rows, bf16
    softmax_row_kernel<<<dim3(N), blk, 0, stream>>>(S, P, N);
    // 5) out = P @ V (f32)
    gemm_wmma<false, 0><<<dim3(D / BN, N / BM), blk, 0, stream>>>(P, Vb, out, N, D, N);
}
